// Model_23175643530014
// MI455X (gfx1250) — matrix operand order repaired and checked by reading, not compiled, not run on hardware
//
#include <hip/hip_runtime.h>

typedef _Float16 f16;
typedef __attribute__((ext_vector_type(16))) _Float16 v16h;
typedef __attribute__((ext_vector_type(4)))  _Float16 v4h;
typedef __attribute__((ext_vector_type(8)))  float    v8f;
typedef __attribute__((ext_vector_type(4)))  float    v4f;

#define N_NODES 50000
#define N_EDGES 800000
#define IN_CH   128
#define HID_CH  128
#define OUT_CH  64

// Native CDNA5 fp32 atomic add (no return -> STOREcnt, fire-and-forget).
// Device scope so the RMW serializes at L2 across all WGPs.
__device__ inline void atomic_add_f32_dev(float* p, float v) {
    asm volatile("global_atomic_add_f32 %0, %1, off scope:SCOPE_DEV"
                 :
                 : "v"(p), "v"(v)
                 : "memory");
}

// ---------------------------------------------------------------------------
// Swizzle a K x N fp32 weight matrix into f16 B-fragment order for
// v_wmma_f32_16x16x32_f16.  Fragment layout (ISA 7.12.2, 16-bit B 32x16):
//   lane L holds 16 halfs: b[j] = B[kb*32 + (j/8)*16 + (L/16)*8 + j%8][t*16 + L%16]
// Fragments stored contiguously (512 halfs each), frag id = t*(K/32) + kb.
// ---------------------------------------------------------------------------
__global__ void k_swizzle_w(const float* __restrict__ W, f16* __restrict__ Wsw,
                            int K, int N) {
    int idx = blockIdx.x * blockDim.x + threadIdx.x;
    if (idx >= K * N) return;
    int f      = idx >> 9;          // fragment id
    int within = idx & 511;
    int lane   = within >> 4;
    int j      = within & 15;
    int kTiles = K >> 5;            // K/32
    int t      = f / kTiles;        // N-tile
    int kb     = f - t * kTiles;    // K-tile
    int k      = kb * 32 + ((j >> 3) << 4) + ((lane >> 4) << 3) + (j & 7);
    int n      = (t << 4) + (lane & 15);
    Wsw[idx] = (f16)W[k * N + n];
}

// ---------------------------------------------------------------------------
// Load a 16x32 f16 A-fragment from a row-major fp32 matrix.
// A-frag layout (ISA 7.12.2, 16-bit A 16x32): lane L, row = m0 + L%16,
//   a[0..7]  = A[row][kb32 + (L/16)*8 + 0..7]
//   a[8..15] = A[row][kb32 + 16 + (L/16)*8 + 0..7]
// ---------------------------------------------------------------------------
template <bool RELU>
__device__ inline v16h load_a_frag(const float* __restrict__ A, int m0,
                                   int kb32, int ldA) {
    int lane = threadIdx.x & 31;
    int row  = m0 + (lane & 15);
    int hh   = lane >> 4;
    const float* p = A + (size_t)row * ldA + kb32 + hh * 8;
    v4f f0 = *(const v4f*)(p + 0);
    v4f f1 = *(const v4f*)(p + 4);
    v4f f2 = *(const v4f*)(p + 16);
    v4f f3 = *(const v4f*)(p + 20);
    v16h a;
#pragma unroll
    for (int j = 0; j < 4; ++j) {
        float x0 = f0[j], x1 = f1[j], x2 = f2[j], x3 = f3[j];
        if (RELU) {
            x0 = fmaxf(x0, 0.f); x1 = fmaxf(x1, 0.f);
            x2 = fmaxf(x2, 0.f); x3 = fmaxf(x3, 0.f);
        }
        a[j]      = (f16)x0;
        a[4 + j]  = (f16)x1;
        a[8 + j]  = (f16)x2;
        a[12 + j] = (f16)x3;
    }
    return a;
}

// ---------------------------------------------------------------------------
// GEMM1: H[50000,128] (f16 out) = X[50000,128] (f32) * Wsw1 (f16 frags)
// 8 waves per block; wave w owns output columns [w*16, w*16+16).
// ---------------------------------------------------------------------------
__global__ __launch_bounds__(256) void k_gemm1(const float* __restrict__ X,
                                               const f16* __restrict__ Wsw,
                                               f16* __restrict__ H) {
    int lane = threadIdx.x & 31;
    int wave = threadIdx.x >> 5;      // 0..7 -> N tile
    int m0   = blockIdx.x << 4;       // 16 rows / block, 50000 % 16 == 0
    v8f c = {};
#pragma unroll
    for (int kb = 0; kb < 4; ++kb) {
        v16h a = load_a_frag<false>(X, m0, kb * 32, IN_CH);
        v16h b = *(const v16h*)(Wsw + (((wave << 2) + kb) << 9) + (lane << 4));
        c = __builtin_amdgcn_wmma_f32_16x16x32_f16(false, a, false, b,
                                                   (short)0, c, false, false);
    }
    int col   = (wave << 4) + (lane & 15);
    int rbase = m0 + ((lane >> 4) << 3);
#pragma unroll
    for (int r = 0; r < 8; ++r)
        H[(size_t)(rbase + r) * HID_CH + col] = (f16)c[r];
}

// ---------------------------------------------------------------------------
// GEMM2: OUT[50000,64] (f32) = relu(AGG[50000,128]) * Wsw2 + b2
// 4 waves per block.
// ---------------------------------------------------------------------------
__global__ __launch_bounds__(128) void k_gemm2(const float* __restrict__ AGG,
                                               const f16* __restrict__ Wsw,
                                               const float* __restrict__ b2,
                                               float* __restrict__ OUT) {
    int lane = threadIdx.x & 31;
    int wave = threadIdx.x >> 5;      // 0..3 -> N tile
    int m0   = blockIdx.x << 4;
    v8f c = {};
#pragma unroll
    for (int kb = 0; kb < 4; ++kb) {
        v16h a = load_a_frag<true>(AGG, m0, kb * 32, HID_CH);
        v16h b = *(const v16h*)(Wsw + (((wave << 2) + kb) << 9) + (lane << 4));
        c = __builtin_amdgcn_wmma_f32_16x16x32_f16(false, a, false, b,
                                                   (short)0, c, false, false);
    }
    int col   = (wave << 4) + (lane & 15);
    float bias = b2[col];
    int rbase = m0 + ((lane >> 4) << 3);
#pragma unroll
    for (int r = 0; r < 8; ++r)
        OUT[(size_t)(rbase + r) * OUT_CH + col] = c[r] + bias;
}

// --------------------------- degree / norm ---------------------------------
__global__ void k_init_deg(float* deg) {
    int i = blockIdx.x * blockDim.x + threadIdx.x;
    if (i < N_NODES) deg[i] = 1.0f;   // self loop
}
__global__ void k_count(const int* __restrict__ dst, float* deg) {
    int e = blockIdx.x * blockDim.x + threadIdx.x;
    if (e < N_EDGES) atomic_add_f32_dev(&deg[dst[e]], 1.0f);
}
__global__ void k_rsqrt(float* deg) {
    int i = blockIdx.x * blockDim.x + threadIdx.x;
    if (i < N_NODES) deg[i] = rsqrtf(deg[i]);   // deg >= 1 always
}

// agg[i][c] = b1[c] + dinv[i]^2 * h[i][c]   (self-loop + bias folded in)
__global__ void k_agg_init(const f16* __restrict__ H,
                           const float* __restrict__ dinv,
                           const float* __restrict__ b1,
                           float* __restrict__ agg) {
    int idx = blockIdx.x * blockDim.x + threadIdx.x;
    if (idx >= N_NODES * HID_CH) return;
    int i  = idx >> 7;
    int ch = idx & 127;
    float di = dinv[i];
    agg[idx] = b1[ch] + di * di * (float)H[idx];
}

// One wave per edge; lane handles 4 channels (8B f16 gather, 4 f32 atomics).
__global__ __launch_bounds__(256) void k_scatter(const int* __restrict__ src,
                                                 const int* __restrict__ dst,
                                                 const float* __restrict__ dinv,
                                                 const f16* __restrict__ H,
                                                 float* __restrict__ agg) {
    int e = (blockIdx.x * blockDim.x + threadIdx.x) >> 5;
    if (e >= N_EDGES) return;
    int lane = threadIdx.x & 31;
    int s = src[e];
    int d = dst[e];
    float nrm = dinv[s] * dinv[d];
    v4h hv = *(const v4h*)(H + (size_t)s * HID_CH + (lane << 2));
    float* ap = agg + (size_t)d * HID_CH + (lane << 2);
#pragma unroll
    for (int j = 0; j < 4; ++j)
        atomic_add_f32_dev(ap + j, nrm * (float)hv[j]);
}

// ---------------------------------------------------------------------------
extern "C" void kernel_launch(void* const* d_in, const int* in_sizes, int n_in,
                              void* d_out, int out_size, void* d_ws,
                              size_t ws_size, hipStream_t stream) {
    const float* x    = (const float*)d_in[0];
    const int*   edge = (const int*)d_in[1];      // [2, E] flat, int32
    const float* W1   = (const float*)d_in[2];
    const float* b1   = (const float*)d_in[3];
    const float* W2   = (const float*)d_in[4];
    const float* b2   = (const float*)d_in[5];
    float* out = (float*)d_out;

    const int* src = edge;
    const int* dst = edge + N_EDGES;

    // workspace layout (all offsets 256B-aligned)
    char* ws = (char*)d_ws;
    f16*   Wsw1 = (f16*)(ws);                                    // 32 KB
    f16*   Wsw2 = (f16*)(ws + 32 * 1024);                        // 16 KB
    float* deg  = (float*)(ws + 64 * 1024);                      // 200 KB
    f16*   H    = (f16*)(ws + 64 * 1024 + 256 * 1024);           // 12.8 MB
    float* agg  = (float*)(ws + 64 * 1024 + 256 * 1024
                              + (size_t)N_NODES * HID_CH * 2);   // 25.6 MB

    // 1) weight swizzle into WMMA B-fragment order (f16)
    k_swizzle_w<<<(IN_CH * HID_CH + 255) / 256, 256, 0, stream>>>(W1, Wsw1,
                                                                  IN_CH, HID_CH);
    k_swizzle_w<<<(HID_CH * OUT_CH + 255) / 256, 256, 0, stream>>>(W2, Wsw2,
                                                                   HID_CH, OUT_CH);
    // 2) degrees -> dinv
    k_init_deg<<<(N_NODES + 255) / 256, 256, 0, stream>>>(deg);
    k_count<<<N_EDGES / 256, 256, 0, stream>>>(dst, deg);
    k_rsqrt<<<(N_NODES + 255) / 256, 256, 0, stream>>>(deg);
    // 3) h = x @ W1  (f16 out)
    k_gemm1<<<N_NODES / 16, 256, 0, stream>>>(x, Wsw1, H);
    // 4) agg = b1 + dinv^2 * h  (self loops + bias)
    k_agg_init<<<(N_NODES * HID_CH) / 256, 256, 0, stream>>>(H, deg, b1, agg);
    // 5) edge scatter: agg[dst] += dinv[s]*dinv[d] * h[src]
    k_scatter<<<N_EDGES / 8, 256, 0, stream>>>(src, dst, deg, H, agg);
    // 6) out = relu(agg) @ W2 + b2
    k_gemm2<<<N_NODES / 16, 128, 0, stream>>>(agg, Wsw2, b2, out);
}
